// PointNetPP_76854144795246
// MI455X (gfx1250) — matrix operand order repaired and checked by reading, not compiled, not run on hardware
//
#include <hip/hip_runtime.h>
#include <hip/hip_bf16.h>

typedef __attribute__((ext_vector_type(16))) _Float16 v16h;
typedef __attribute__((ext_vector_type(8)))  _Float16 v8h;
typedef __attribute__((ext_vector_type(8)))  float    v8f;

// ---------------------------------------------------------------------------
// Problem constants (reference: B=16, N=8192, 3 SA stages, 4 MLP layers each)
// ---------------------------------------------------------------------------
#define BATCH 16

// Real (reference) layer dims per stage, and padded dims (Cin padded to x32
// so WMMA K-chunks are whole; padding channels are exact zeros end-to-end).
static const int RD[3][5] = {
  {  3,   3,  64,  64,  128},
  {131, 128, 128, 128,  256},
  {259, 256, 256, 512, 1024},
};
static const int PD[3][5] = {
  { 32,  32,  64,  64,  128},
  {160, 128, 128, 128,  256},
  {288, 256, 256, 512, 1024},
};

struct MlpArgs {
  int dims[5];          // padded dims
  int cout_last;        // real final channel count
  int b0cols, b1cols;   // LDS ping-pong buffer widths (halfwords per row)
  long long woff[4];    // per-layer offset (in halfwords) into packed f16 weights
  long long boff[4];    // per-layer offset (in floats) into packed f32 biases
};

// ---------------------------------------------------------------------------
// Weight prep: fp32 [Cin][Cout] -> padded transposed f16 [Coutp][Cinp], plus
// zero-padded f32 bias. Weights are tiny; they stay resident in L2.
// ---------------------------------------------------------------------------
__global__ void prep_weights_kernel(const float* __restrict__ w,
                                    const float* __restrict__ b,
                                    int Cin, int Cout, int Cinp, int Coutp,
                                    _Float16* __restrict__ wt,
                                    float* __restrict__ bout) {
  long long i = (long long)blockIdx.x * blockDim.x + threadIdx.x;
  long long tot = (long long)Coutp * Cinp;
  if (i < tot) {
    int co = (int)(i / Cinp);
    int ci = (int)(i % Cinp);
    float v = (ci < Cin && co < Cout) ? w[(size_t)ci * Cout + co] : 0.0f;
    wt[i] = (_Float16)v;
  }
  if (i < Coutp) bout[i] = (i < Cout) ? b[i] : 0.0f;
}

// ---------------------------------------------------------------------------
// Farthest point sampling. One block per batch. dists[] lives in LDS.
// Matches jnp.argmax first-index tie-breaking. Emits new_xyz directly.
// ---------------------------------------------------------------------------
__global__ void fps_kernel(const float* __restrict__ xyz,
                           float* __restrict__ new_xyz,
                           int N, int S) {
  extern __shared__ float fps_smem[];
  float* dists = fps_smem;                       // N floats
  float* sval  = fps_smem + N;                   // blockDim floats
  int*   sidx  = (int*)(sval + blockDim.x);      // blockDim ints

  const int b   = blockIdx.x;
  const int tid = threadIdx.x;
  const float* px = xyz + (size_t)b * N * 3;

  for (int p = tid; p < N; p += blockDim.x) dists[p] = 1e10f;
  __syncthreads();

  int far = 0;
  for (int s = 0; s < S; ++s) {
    float cx = px[far * 3 + 0];
    float cy = px[far * 3 + 1];
    float cz = px[far * 3 + 2];
    if (tid == 0) {
      float* q = new_xyz + ((size_t)b * S + s) * 3;
      q[0] = cx; q[1] = cy; q[2] = cz;
    }
    float bv = -1.0f;
    int   bi = 0x7fffffff;
    for (int p = tid; p < N; p += blockDim.x) {
      float dx = px[p * 3 + 0] - cx;
      float dy = px[p * 3 + 1] - cy;
      float dz = px[p * 3 + 2] - cz;
      float d  = dx * dx + dy * dy + dz * dz;
      float nd = fminf(dists[p], d);
      dists[p] = nd;
      if (nd > bv) { bv = nd; bi = p; }   // strided ascending p => first-index kept
    }
    sval[tid] = bv; sidx[tid] = bi;
    __syncthreads();
    for (int off = blockDim.x >> 1; off > 0; off >>= 1) {
      if (tid < off) {
        float ov = sval[tid + off]; int oi = sidx[tid + off];
        if (ov > sval[tid] || (ov == sval[tid] && oi < sidx[tid])) {
          sval[tid] = ov; sidx[tid] = oi;
        }
      }
      __syncthreads();
    }
    far = sidx[0];
    __syncthreads();
  }
}

// ---------------------------------------------------------------------------
// Ball query: one wave32 per (b,s). Chunked ballot scan keeps the first K
// in-range point indices in ascending index order; pads with the first hit.
// ---------------------------------------------------------------------------
__global__ void ball_query_kernel(const float* __restrict__ xyz,
                                  const float* __restrict__ new_xyz,
                                  int* __restrict__ idx,
                                  int N, int S, int K, float r2) {
  int gw   = (int)((blockIdx.x * (long long)blockDim.x + threadIdx.x) >> 5);
  int lane = threadIdx.x & 31;
  if (gw >= BATCH * S) return;
  int b = gw / S, s = gw % S;

  const float* q = new_xyz + ((size_t)b * S + s) * 3;
  float qx = q[0], qy = q[1], qz = q[2];
  const float* px = xyz + (size_t)b * N * 3;
  int* out = idx + ((size_t)b * S + s) * K;

  int found = 0;
  for (int base = 0; base < N && found < K; base += 32) {
    int p = base + lane;
    bool valid = false;
    if (p < N) {
      float dx = px[p * 3 + 0] - qx;
      float dy = px[p * 3 + 1] - qy;
      float dz = px[p * 3 + 2] - qz;
      valid = (dx * dx + dy * dy + dz * dz) < r2;
    }
    unsigned mask = (unsigned)__ballot(valid);
    int pos = found + __popc(mask & ((1u << lane) - 1u));
    if (valid && pos < K) out[pos] = p;
    found += __popc(mask);
  }
  if (found > K) found = K;
  if (found < K) {
    int first = (found > 0) ? out[0] : 0;
    for (int j = found + lane; j < K; j += 32) out[j] = first;
  }
}

// ---------------------------------------------------------------------------
// Grouped shared-MLP + max-pool. One 128-thread workgroup per (b,s) group.
// X0 = [relative xyz | gathered features | zero pad] as f16 in LDS; four
// layers of v_wmma_f32_16x16x32_f16 ping-ponging LDS buffers; last layer
// fuses ReLU + max-over-K via LDS atomicMax on the int pattern (ReLU >= 0).
// ---------------------------------------------------------------------------
__global__ __launch_bounds__(128)
void sa_mlp_kernel(const float* __restrict__ xyz,
                   const float* __restrict__ feat, int Cfeat,
                   const float* __restrict__ new_xyz,
                   const int* __restrict__ idx,
                   const _Float16* __restrict__ wt,
                   const float* __restrict__ bias,
                   float* __restrict__ fout,
                   int N, int S, int K, MlpArgs args) {
  extern __shared__ char mlp_smem[];
  _Float16* buf0   = (_Float16*)mlp_smem;
  _Float16* buf1   = buf0 + (size_t)K * args.b0cols;
  float*    outmax = (float*)(buf1 + (size_t)K * args.b1cols);

  const int b    = blockIdx.x / S;
  const int s    = blockIdx.x % S;
  const int tid  = threadIdx.x;
  const int lane = tid & 31;
  const int wave = tid >> 5;
  const int nwav = blockDim.x >> 5;

  const int*   gidx = idx + ((size_t)b * S + s) * K;
  const float* q    = new_xyz + ((size_t)b * S + s) * 3;

  // ---- build X0 in LDS (f16, tight ld = dims[0]) ----
  const int C0 = args.dims[0];
  for (int f = tid; f < K * C0; f += blockDim.x) {
    int k = f / C0, c = f % C0;
    int p = gidx[k];
    float v = 0.0f;
    if (c < 3)                v = xyz[((size_t)b * N + p) * 3 + c] - q[c];
    else if (c - 3 < Cfeat)   v = feat[((size_t)b * N + p) * Cfeat + (c - 3)];
    buf0[(size_t)k * C0 + c] = (_Float16)v;
  }
  for (int c = tid; c < args.dims[4]; c += blockDim.x) outmax[c] = 0.0f;
  __syncthreads();

  _Float16* bufs[2] = {buf0, buf1};
  const int half = lane >> 4;    // lane group: selects K-offset in A/B frags
  const int nn   = lane & 15;    // A row / B-&-D column within 16x16 tile

#pragma unroll
  for (int l = 0; l < 4; ++l) {
    const int Cin  = args.dims[l];
    const int Cout = args.dims[l + 1];
    const _Float16* W  = wt + args.woff[l];     // transposed [Cout][Cin]
    const float*    Bv = bias + args.boff[l];
    _Float16* X = bufs[l & 1];
    _Float16* Y = bufs[(l + 1) & 1];
    const bool last = (l == 3);

    const int mt_n = K >> 4, nt_n = Cout >> 4;
    const int tiles = mt_n * nt_n;

    for (int t = wave; t < tiles; t += nwav) {
      int mt = t % mt_n, nt = t / mt_n;
      v8f acc = {};
      const int row = mt * 16 + nn;
      for (int kc = 0; kc < Cin; kc += 32) {
        // A fragment: 16-bit A 16x32 layout -> two contiguous 16B LDS loads
        const _Float16* ap = X + (size_t)row * Cin + kc + half * 8;
        v8h alo = *(const v8h*)ap;
        v8h ahi = *(const v8h*)(ap + 16);
        v16h a;
#pragma unroll
        for (int i = 0; i < 8; ++i) { a[i] = alo[i]; a[i + 8] = ahi[i]; }
        // B fragment: 32x16, same K order as A: lanes 0-15 K=0..7,16..23 / lanes 16-31 K=8..15,24..31;
        // transposed weights make this two contiguous 8-halfword runs.
        const _Float16* bp = W + (size_t)(nt * 16 + nn) * Cin + kc + half * 8;
        v8h blo = *(const v8h*)bp;
        v8h bhi = *(const v8h*)(bp + 16);
        v16h bm;
#pragma unroll
        for (int i = 0; i < 8; ++i) { bm[i] = blo[i]; bm[i + 8] = bhi[i]; }

        acc = __builtin_amdgcn_wmma_f32_16x16x32_f16(
            false, a, false, bm, (short)0, acc, false, false);
      }
      // epilogue: C/D layout -> lane holds rows mt*16 + half*8 + i, col nn
      const int   col  = nt * 16 + nn;
      const float bcol = Bv[col];
      if (!last) {
#pragma unroll
        for (int i = 0; i < 8; ++i) {
          int r = mt * 16 + half * 8 + i;
          float v = acc[i] + bcol;
          v = v > 0.0f ? v : 0.0f;
          Y[(size_t)r * Cout + col] = (_Float16)v;
        }
      } else {
        float vmax = 0.0f;
#pragma unroll
        for (int i = 0; i < 8; ++i) {
          float v = acc[i] + bcol;
          v = v > 0.0f ? v : 0.0f;
          vmax = fmaxf(vmax, v);
        }
        // ReLU output >= 0 -> int bit pattern is order-preserving
        atomicMax((int*)&outmax[col], __float_as_int(vmax));
      }
    }
    __syncthreads();
  }

  for (int c = tid; c < args.cout_last; c += blockDim.x)
    fout[((size_t)b * S + s) * args.cout_last + c] = outmax[c];
}

// ---------------------------------------------------------------------------
// Final output: d_out = [xyz3 (16,32,3) | max_s f3 (16,1024)]
// ---------------------------------------------------------------------------
__global__ void finalize_kernel(const float* __restrict__ new_xyz3,
                                const float* __restrict__ f3,
                                float* __restrict__ out) {
  int i = blockIdx.x * blockDim.x + threadIdx.x;
  if (i < BATCH * 32 * 3) out[i] = new_xyz3[i];
  if (i < BATCH * 1024) {
    int b = i / 1024, c = i % 1024;
    float m = 0.0f;  // ReLU outputs are >= 0
    for (int s = 0; s < 32; ++s)
      m = fmaxf(m, f3[((size_t)b * 32 + s) * 1024 + c]);
    out[BATCH * 32 * 3 + i] = m;
  }
}

// ---------------------------------------------------------------------------
// Host orchestration
// ---------------------------------------------------------------------------
extern "C" void kernel_launch(void* const* d_in, const int* in_sizes, int n_in,
                              void* d_out, int out_size, void* d_ws, size_t ws_size,
                              hipStream_t stream) {
  (void)in_sizes; (void)n_in; (void)out_size; (void)ws_size;
  const float* xyz = (const float*)d_in[0];

  // ---- carve workspace ----
  char* ws = (char*)d_ws;
  size_t off = 0;
  auto carve = [&](size_t bytes) -> char* {
    char* p = ws + off;
    off = (off + bytes + 255) & ~(size_t)255;
    return p;
  };

  long long woff[3][4], boff[3][4];
  size_t wtot = 0, btot = 0;
  for (int v = 0; v < 3; ++v)
    for (int l = 0; l < 4; ++l) {
      woff[v][l] = (long long)wtot;
      wtot += (size_t)PD[v][l] * PD[v][l + 1];
      boff[v][l] = (long long)btot;
      btot += (size_t)PD[v][l + 1];
    }

  _Float16* wt   = (_Float16*)carve(wtot * sizeof(_Float16));
  float*    bias = (float*)carve(btot * sizeof(float));
  float* nx1 = (float*)carve((size_t)BATCH * 512 * 3 * 4);
  float* nx2 = (float*)carve((size_t)BATCH * 128 * 3 * 4);
  float* nx3 = (float*)carve((size_t)BATCH * 32  * 3 * 4);
  int* idx1 = (int*)carve((size_t)BATCH * 512 * 32  * 4);
  int* idx2 = (int*)carve((size_t)BATCH * 128 * 64  * 4);
  int* idx3 = (int*)carve((size_t)BATCH * 32  * 128 * 4);
  float* f1 = (float*)carve((size_t)BATCH * 512 * 128  * 4);
  float* f2 = (float*)carve((size_t)BATCH * 128 * 256  * 4);
  float* f3 = (float*)carve((size_t)BATCH * 32  * 1024 * 4);

  // ---- weight prep (fp32 -> padded transposed f16) ----
  for (int v = 0; v < 3; ++v)
    for (int l = 0; l < 4; ++l) {
      const float* w  = (const float*)d_in[1 + v * 8 + l * 2];
      const float* bb = (const float*)d_in[2 + v * 8 + l * 2];
      int Cin = RD[v][l], Cout = RD[v][l + 1];
      int Cinp = PD[v][l], Coutp = PD[v][l + 1];
      long long tot = (long long)Cinp * Coutp;
      int blk = (int)((tot + 255) / 256);
      prep_weights_kernel<<<dim3(blk), dim3(256), 0, stream>>>(
          w, bb, Cin, Cout, Cinp, Coutp, wt + woff[v][l], bias + boff[v][l]);
    }

  // ---- per-stage config ----
  const int   Ns[3]  = {8192, 512, 128};
  const int   Ss[3]  = {512, 128, 32};
  const int   Ks[3]  = {32, 64, 128};
  const float R2[3]  = {0.04f, 0.16f, 0.64f};
  const int   CF[3]  = {0, 128, 256};
  const float* xsrc[3] = {xyz, nx1, nx2};
  const float* fsrc[3] = {nullptr, f1, f2};
  float* nxo[3] = {nx1, nx2, nx3};
  int*   ido[3] = {idx1, idx2, idx3};
  float* fo[3]  = {f1, f2, f3};

  for (int v = 0; v < 3; ++v) {
    const int N = Ns[v], S = Ss[v], K = Ks[v];

    // FPS
    size_t fps_sh = (size_t)(N + 2 * 256) * 4;
    fps_kernel<<<dim3(BATCH), dim3(256), fps_sh, stream>>>(xsrc[v], nxo[v], N, S);

    // Ball query: one wave per (b,s)
    int waves = BATCH * S;
    int bq_blocks = (waves * 32 + 255) / 256;
    ball_query_kernel<<<dim3(bq_blocks), dim3(256), 0, stream>>>(
        xsrc[v], nxo[v], ido[v], N, S, K, R2[v]);

    // Grouped MLP + max-pool (WMMA)
    MlpArgs a;
    for (int d = 0; d < 5; ++d) a.dims[d] = PD[v][d];
    a.cout_last = RD[v][4];
    int c02 = PD[v][0] > PD[v][2] ? PD[v][0] : PD[v][2];
    int c13 = PD[v][1] > PD[v][3] ? PD[v][1] : PD[v][3];
    a.b0cols = c02;
    a.b1cols = c13;
    for (int l = 0; l < 4; ++l) { a.woff[l] = woff[v][l]; a.boff[l] = boff[v][l]; }
    size_t mlp_sh = (size_t)K * (c02 + c13) * sizeof(_Float16)
                  + (size_t)PD[v][4] * sizeof(float);
    hipFuncSetAttribute(reinterpret_cast<const void*>(sa_mlp_kernel),
                        hipFuncAttributeMaxDynamicSharedMemorySize,
                        (int)mlp_sh);
    sa_mlp_kernel<<<dim3(BATCH * S), dim3(128), mlp_sh, stream>>>(
        xsrc[v], fsrc[v], CF[v], nxo[v], ido[v], wt, bias, fo[v], N, S, K, a);
  }

  // ---- final output assembly ----
  finalize_kernel<<<dim3((BATCH * 1024 + 255) / 256), dim3(256), 0, stream>>>(
      nx3, f3, (float*)d_out);
}
